// AttentionWithFeedForward_73151882986045
// MI455X (gfx1250) — matrix operand order repaired and checked by reading, not compiled, not run on hardware
//
#include <hip/hip_runtime.h>
#include <math.h>

typedef __bf16 bf16_t;
typedef __attribute__((ext_vector_type(16))) __bf16 v16bf;
typedef __attribute__((ext_vector_type(8)))  __bf16 v8bf;
typedef __attribute__((ext_vector_type(8)))  float  v8f;
typedef int v4i32 __attribute__((vector_size(16)));   // matches builtin V4i

#define AS1 __attribute__((address_space(1)))
#define AS3 __attribute__((address_space(3)))

#if __has_builtin(__builtin_amdgcn_global_load_async_to_lds_b128)
#define HAVE_ASYNC_COPY 1
#else
#define HAVE_ASYNC_COPY 0
#endif

// ---------------------------------------------------------------- helpers ---

static __device__ __forceinline__ v8bf zero_v8bf() {
  v8bf z;
#pragma unroll
  for (int i = 0; i < 8; ++i) z[i] = (bf16_t)0.0f;
  return z;
}

static __device__ __forceinline__ v16bf make_v16(v8bf lo, v8bf hi) {
  v16bf r;
#pragma unroll
  for (int i = 0; i < 8; ++i) { r[i] = lo[i]; r[i + 8] = hi[i]; }
  return r;
}

// 16-byte global -> LDS copy: async (ASYNCcnt-tracked, no VGPR round trip)
// when the gfx1250 builtin exists, else synchronous fallback.
static __device__ __forceinline__ void copy16_g2l(const bf16_t* g, bf16_t* l) {
#if HAVE_ASYNC_COPY
  __builtin_amdgcn_global_load_async_to_lds_b128(
      (AS1 v4i32*)const_cast<bf16_t*>(g), (AS3 v4i32*)l, 0, 0);
#else
  *(v8bf*)l = *(const v8bf*)g;
#endif
}

static __device__ __forceinline__ void wait_async_copies() {
#if HAVE_ASYNC_COPY
#if __has_builtin(__builtin_amdgcn_s_wait_asynccnt)
  __builtin_amdgcn_s_wait_asynccnt(0);
#else
  asm volatile("s_wait_asynccnt 0" ::: "memory");
#endif
#endif
}

// A-fragment (16x32 bf16) per CDNA5 ISA 7.12.2:
//  lane l<16 : row=l,    K=0..7 (v0..3), K=16..23 (v4..7)
//  lane l>=16: row=l-16, K=8..15,        K=24..31
static __device__ __forceinline__ v16bf load_a_frag(const bf16_t* base, int ld) {
  const int l = threadIdx.x & 31;
  const bf16_t* p = base + (size_t)(l & 15) * ld + ((l >> 4) << 3);
  return make_v16(*(const v8bf*)p, *(const v8bf*)(p + 16));
}

// B-fragment (32x16 bf16): lane l holds column N=l%16, K in same order as A-fragment.
static __device__ __forceinline__ v16bf load_b_frag(const bf16_t* base, int ld) {
  const int l = threadIdx.x & 31;
  const bf16_t* p = base + (l & 15) + (size_t)((l >> 4) << 3) * ld;
  v16bf r;
#pragma unroll
  for (int i = 0; i < 8; ++i) { r[i] = p[(size_t)i * ld]; r[i + 8] = p[(size_t)(16 + i) * ld]; }
  return r;
}

static __device__ __forceinline__ v8f wmma_bf16(v16bf a, v16bf b, v8f c) {
  return __builtin_amdgcn_wmma_f32_16x16x32_bf16(false, a, false, b, (short)0, c,
                                                 false, false);
}

static __device__ __forceinline__ float gelu_tanh(float x) {
  float x3 = x * x * x;
  return 0.5f * x * (1.0f + tanhf(0.7978845608028654f * (x + 0.044715f * x3)));
}

// ------------------------------------------------------------ cvt kernel ---

__global__ void cvt_f32_bf16(const float* __restrict__ in,
                             bf16_t* __restrict__ out, long long n) {
  long long i = (long long)blockIdx.x * blockDim.x + threadIdx.x;
  long long stride = (long long)gridDim.x * blockDim.x;
  for (; i < n; i += stride) out[i] = (bf16_t)in[i];
}

// ------------------------------------------------------- WMMA GEMM kernel ---
// C[M,N] = act(A[M,K] @ B[K,N] + bias). 128x128 tile, BK=32, 8 waves (4x2),
// double-buffered LDS with async global->LDS staging overlapped with WMMA.
__global__ __launch_bounds__(256)
void gemm_bf16_kernel(const bf16_t* __restrict__ A, const bf16_t* __restrict__ B,
                      const float* __restrict__ bias,
                      float* __restrict__ outF, bf16_t* __restrict__ outB,
                      int M, int N, int K, int act_gelu) {
  __shared__ __attribute__((aligned(16))) bf16_t As[2][128][40];
  __shared__ __attribute__((aligned(16))) bf16_t Bs[2][32][136];

  const int tid = threadIdx.x;
  const int l = tid & 31;
  const int wave = tid >> 5;
  const int wm = wave >> 1;   // 0..3 (32 rows each)
  const int wn = wave & 1;    // 0..1 (64 cols each)
  const int m0 = blockIdx.y * 128;
  const int n0 = blockIdx.x * 128;

  // issue all copies for one 128x32 A tile + 32x128 B tile into buffer `buf`
  auto stage = [&](int buf, int k0) {
#pragma unroll
    for (int it = 0; it < 2; ++it) {           // A: 128x32
      int idx = tid + it * 256;
      int r = idx >> 2;
      int c = (idx & 3) << 3;
      int gr = m0 + r;
      if (gr < M)
        copy16_g2l(A + (size_t)gr * K + k0 + c, &As[buf][r][c]);
      else
        *(v8bf*)&As[buf][r][c] = zero_v8bf();
    }
#pragma unroll
    for (int it = 0; it < 2; ++it) {           // B: 32x128 (never OOB)
      int idx = tid + it * 256;
      int r = idx >> 4;
      int c = (idx & 15) << 3;
      copy16_g2l(B + (size_t)(k0 + r) * N + n0 + c, &Bs[buf][r][c]);
    }
  };

  v8f acc[2][4];
#pragma unroll
  for (int i = 0; i < 2; ++i)
#pragma unroll
    for (int j = 0; j < 4; ++j)
#pragma unroll
      for (int r = 0; r < 8; ++r) acc[i][j][r] = 0.0f;

  const int nk = K >> 5;
  stage(0, 0);
  wait_async_copies();
  __syncthreads();

  for (int ik = 0; ik < nk; ++ik) {
    const int buf = ik & 1;
    if (ik + 1 < nk) stage(buf ^ 1, (ik + 1) << 5);   // prefetch next tile

    v16bf af0 = load_a_frag(&As[buf][wm * 32][0], 40);
    v16bf af1 = load_a_frag(&As[buf][wm * 32 + 16][0], 40);
#pragma unroll
    for (int jn = 0; jn < 4; ++jn) {
      v16bf bf = load_b_frag(&Bs[buf][0][wn * 64 + jn * 16], 136);
      acc[0][jn] = wmma_bf16(af0, bf, acc[0][jn]);
      acc[1][jn] = wmma_bf16(af1, bf, acc[1][jn]);
    }
    wait_async_copies();
    __syncthreads();
  }

  const int half = (l >> 4) << 3;
  const int cc = l & 15;
#pragma unroll
  for (int im = 0; im < 2; ++im)
#pragma unroll
    for (int jn = 0; jn < 4; ++jn)
#pragma unroll
      for (int r = 0; r < 8; ++r) {
        int row = wm * 32 + im * 16 + r + half;
        int col = wn * 64 + jn * 16 + cc;
        int gr = m0 + row, gc = n0 + col;
        if (gr < M) {
          float v = acc[im][jn][r] + bias[gc];
          if (act_gelu) v = gelu_tanh(v);
          if (outF) outF[(size_t)gr * N + gc] = v;
          if (outB) outB[(size_t)gr * N + gc] = (bf16_t)v;
        }
      }
}

// --------------------------------------------------- flash attention kernel --
// d_head=64. Block: 4 waves x 16 query rows = 64 rows. 32-key blocks.
__global__ __launch_bounds__(128)
void flash_attn_kernel(const bf16_t* __restrict__ Q, int qs,
                       const bf16_t* __restrict__ Kp,
                       const bf16_t* __restrict__ Vp, int kvs,
                       bf16_t* __restrict__ O, int os,
                       int Sq, int Skv, int nheads, float scale) {
  __shared__ __attribute__((aligned(16))) bf16_t KT[64][40];       // [d][key]
  __shared__ __attribute__((aligned(16))) bf16_t Vs[32][72];       // [key][d]
  __shared__ __attribute__((aligned(16))) bf16_t Ps[4][16][32];    // per-wave P

  const int tid = threadIdx.x;
  const int l = tid & 31;
  const int w = tid >> 5;
  const int bh = blockIdx.y;
  const int b = bh / nheads, h = bh % nheads;

  const bf16_t* qbase = Q + (size_t)b * Sq * qs + h * 64;
  const bf16_t* kbase = Kp + (size_t)b * Skv * kvs + h * 64;
  const bf16_t* vbase = Vp + (size_t)b * Skv * kvs + h * 64;
  bf16_t* obase = O + (size_t)b * Sq * os + h * 64;

  const int q0 = blockIdx.x * 64 + w * 16;

  v16bf aq0 = load_a_frag(qbase + (size_t)q0 * qs, qs);
  v16bf aq1 = load_a_frag(qbase + (size_t)q0 * qs + 32, qs);

  float m[8], sm[8];
#pragma unroll
  for (int r = 0; r < 8; ++r) { m[r] = -1e30f; sm[r] = 0.0f; }
  v8f o[4];
#pragma unroll
  for (int n = 0; n < 4; ++n)
#pragma unroll
    for (int r = 0; r < 8; ++r) o[n][r] = 0.0f;

  const int kk = tid & 31;
  const int db = (tid >> 5) << 4;   // 0,16,32,48
  const int half = (l >> 4) << 3;
  const int cc = l & 15;

  for (int j0 = 0; j0 < Skv; j0 += 32) {
    // ---- stage K (transposed, sync) and V (async) -----------------------
    {
      bool valid = (j0 + kk) < Skv;
      v8bf ka, kb;
      if (valid) {
        const bf16_t* kp = kbase + (size_t)(j0 + kk) * kvs + db;
        ka = *(const v8bf*)kp; kb = *(const v8bf*)(kp + 8);
        const bf16_t* vp = vbase + (size_t)(j0 + kk) * kvs + db;
        copy16_g2l(vp, &Vs[kk][db]);
        copy16_g2l(vp + 8, &Vs[kk][db + 8]);
      } else {
        ka = zero_v8bf(); kb = ka;
        // keep masked-out V rows at 0 so 0*NaN can't leak into the PV WMMA
        *(v8bf*)&Vs[kk][db] = ka;
        *(v8bf*)&Vs[kk][db + 8] = ka;
      }
#pragma unroll
      for (int i = 0; i < 8; ++i) {
        KT[db + i][kk] = ka[i];
        KT[db + 8 + i][kk] = kb[i];
      }
    }
    wait_async_copies();
    __syncthreads();

    // ---- S = scale * Q K^T, masked --------------------------------------
    v8f s[2];
#pragma unroll
    for (int nb = 0; nb < 2; ++nb) {
      v8f c;
#pragma unroll
      for (int r = 0; r < 8; ++r) c[r] = 0.0f;
      c = wmma_bf16(aq0, load_b_frag(&KT[0][nb * 16], 40), c);
      c = wmma_bf16(aq1, load_b_frag(&KT[32][nb * 16], 40), c);
      bool oob = (j0 + nb * 16 + cc) >= Skv;
#pragma unroll
      for (int r = 0; r < 8; ++r) s[nb][r] = oob ? -1e30f : c[r] * scale;
    }

    // ---- online softmax (rows striped over 16-lane halves) --------------
    float p0[8], p1[8], corr[8];
#pragma unroll
    for (int r = 0; r < 8; ++r) {
      float mx = fmaxf(s[0][r], s[1][r]);
      mx = fmaxf(mx, __shfl_xor(mx, 1, 32));
      mx = fmaxf(mx, __shfl_xor(mx, 2, 32));
      mx = fmaxf(mx, __shfl_xor(mx, 4, 32));
      mx = fmaxf(mx, __shfl_xor(mx, 8, 32));
      float mn = fmaxf(m[r], mx);
      corr[r] = __expf(m[r] - mn);
      float e0 = __expf(s[0][r] - mn);
      float e1 = __expf(s[1][r] - mn);
      float rs = e0 + e1;
      rs += __shfl_xor(rs, 1, 32);
      rs += __shfl_xor(rs, 2, 32);
      rs += __shfl_xor(rs, 4, 32);
      rs += __shfl_xor(rs, 8, 32);
      sm[r] = sm[r] * corr[r] + rs;
      m[r] = mn;
      p0[r] = e0; p1[r] = e1;
    }
#pragma unroll
    for (int n = 0; n < 4; ++n)
#pragma unroll
      for (int r = 0; r < 8; ++r) o[n][r] *= corr[r];

    // ---- P: C-layout -> A-layout through per-wave LDS -------------------
#pragma unroll
    for (int r = 0; r < 8; ++r) {
      Ps[w][r + half][cc] = (bf16_t)p0[r];
      Ps[w][r + half][16 + cc] = (bf16_t)p1[r];
    }
    asm volatile("s_wait_dscnt 0" ::: "memory");  // same-wave DS ordering
    v16bf ap = load_a_frag(&Ps[w][0][0], 32);

    // ---- O += P V -------------------------------------------------------
#pragma unroll
    for (int n = 0; n < 4; ++n)
      o[n] = wmma_bf16(ap, load_b_frag(&Vs[0][n * 16], 72), o[n]);

    __syncthreads();
  }

  // ---- normalize + store (bf16) -----------------------------------------
#pragma unroll
  for (int n = 0; n < 4; ++n)
#pragma unroll
    for (int r = 0; r < 8; ++r) {
      float v = o[n][r] / sm[r];
      obase[(size_t)(q0 + r + half) * os + n * 16 + cc] = (bf16_t)v;
    }
}

// ---------------------------------------------- residual + layernorm (512) --

__global__ __launch_bounds__(256)
void residual_ln_kernel(const float* __restrict__ X, const float* __restrict__ D,
                        const float* __restrict__ G, const float* __restrict__ Bt,
                        float* __restrict__ Y, bf16_t* __restrict__ Yb) {
  __shared__ float red[16];
  const size_t base = (size_t)blockIdx.x * 512;
  const int tid = threadIdx.x;
  const int i0 = tid * 2;

  float v0 = X[base + i0] + D[base + i0];
  float v1 = X[base + i0 + 1] + D[base + i0 + 1];
  float s = v0 + v1;
  float q = v0 * v0 + v1 * v1;
#pragma unroll
  for (int off = 16; off >= 1; off >>= 1) {
    s += __shfl_xor(s, off, 32);
    q += __shfl_xor(q, off, 32);
  }
  const int wv = tid >> 5;
  if ((tid & 31) == 0) { red[wv] = s; red[8 + wv] = q; }
  __syncthreads();
  float S = 0.0f, Q2 = 0.0f;
#pragma unroll
  for (int k = 0; k < 8; ++k) { S += red[k]; Q2 += red[8 + k]; }
  const float mu = S * (1.0f / 512.0f);
  const float var = Q2 * (1.0f / 512.0f) - mu * mu;
  const float rs = rsqrtf(var + 1e-5f);
  float o0 = (v0 - mu) * rs * G[i0] + Bt[i0];
  float o1 = (v1 - mu) * rs * G[i0 + 1] + Bt[i0 + 1];
  Y[base + i0] = o0;
  Y[base + i0 + 1] = o1;
  if (Yb) {
    Yb[base + i0] = (bf16_t)o0;
    Yb[base + i0 + 1] = (bf16_t)o1;
  }
}

// -------------------------------------------------------------- launcher ---

extern "C" void kernel_launch(void* const* d_in, const int* in_sizes, int n_in,
                              void* d_out, int out_size, void* d_ws, size_t ws_size,
                              hipStream_t stream) {
  (void)in_sizes; (void)n_in; (void)out_size; (void)ws_size;

  const float* x       = (const float*)d_in[0];
  const float* y       = (const float*)d_in[1];
  const float* sa_in_w = (const float*)d_in[2];
  const float* sa_in_b = (const float*)d_in[3];
  const float* sa_ow   = (const float*)d_in[4];
  const float* sa_ob   = (const float*)d_in[5];
  const float* ca_qw   = (const float*)d_in[6];
  const float* ca_qb   = (const float*)d_in[7];
  const float* ca_kw   = (const float*)d_in[8];
  const float* ca_kb   = (const float*)d_in[9];
  const float* ca_vw   = (const float*)d_in[10];
  const float* ca_vb   = (const float*)d_in[11];
  const float* ca_ow   = (const float*)d_in[12];
  const float* ca_ob   = (const float*)d_in[13];
  const float* ff_w1   = (const float*)d_in[14];
  const float* ff_b1   = (const float*)d_in[15];
  const float* ff_w2   = (const float*)d_in[16];
  const float* ff_b2   = (const float*)d_in[17];
  const float* ln1_g   = (const float*)d_in[18];
  const float* ln1_b   = (const float*)d_in[19];
  const float* ln2_g   = (const float*)d_in[20];
  const float* ln2_b   = (const float*)d_in[21];
  const float* ln3_g   = (const float*)d_in[22];
  const float* ln3_b   = (const float*)d_in[23];
  float* out = (float*)d_out;

  const int B = 2, S = 4096, D = 512, DC = 768, FF = 2048, H = 8;
  const int M  = B * S;     // 8192 tokens
  const int MY = B * 77;    // 154 cross tokens

  char* ws = (char*)d_ws;
  size_t off = 0;
  auto alloc = [&](size_t bytes) -> void* {
    void* p = ws + off;
    off += (bytes + 255) & ~(size_t)255;
    return p;
  };

  // qkv + attn first so the FFN hidden buffer can alias them exactly.
  bf16_t* qkv  = (bf16_t*)alloc((size_t)M * 3 * D * 2);   // 25,165,824 B
  bf16_t* attn = (bf16_t*)alloc((size_t)M * D * 2);       //  8,388,608 B
  bf16_t* ffh  = qkv;  // M*FF*2 = 33,554,432 B == qkv+attn (both dead by then)

  bf16_t* xb   = (bf16_t*)alloc((size_t)M * D * 2);
  bf16_t* yb   = (bf16_t*)alloc((size_t)MY * DC * 2);
  bf16_t* w_in = (bf16_t*)alloc((size_t)D * 3 * D * 2);
  bf16_t* w_so = (bf16_t*)alloc((size_t)D * D * 2);
  bf16_t* w_cq = (bf16_t*)alloc((size_t)D * D * 2);
  bf16_t* w_ck = (bf16_t*)alloc((size_t)DC * D * 2);
  bf16_t* w_cv = (bf16_t*)alloc((size_t)DC * D * 2);
  bf16_t* w_co = (bf16_t*)alloc((size_t)D * D * 2);
  bf16_t* w_f1 = (bf16_t*)alloc((size_t)D * FF * 2);
  bf16_t* w_f2 = (bf16_t*)alloc((size_t)FF * D * 2);
  bf16_t* qc   = (bf16_t*)alloc((size_t)M * D * 2);
  bf16_t* kc   = (bf16_t*)alloc((size_t)MY * D * 2);
  bf16_t* vc   = (bf16_t*)alloc((size_t)MY * D * 2);
  bf16_t* x1b  = (bf16_t*)alloc((size_t)M * D * 2);
  bf16_t* x2b  = (bf16_t*)alloc((size_t)M * D * 2);
  float*  proj = (float*)alloc((size_t)M * D * 4);
  float*  x1f  = (float*)alloc((size_t)M * D * 4);
  float*  x2f  = (float*)alloc((size_t)M * D * 4);

  auto cvt = [&](const float* src, bf16_t* dst, long long n) {
    int blocks = (int)((n + 1023) / 1024);
    if (blocks > 2048) blocks = 2048;
    cvt_f32_bf16<<<blocks, 256, 0, stream>>>(src, dst, n);
  };
  auto gemm = [&](const bf16_t* A, const bf16_t* Bm, const float* bias,
                  float* oF, bf16_t* oB, int Mm, int Nn, int Kk, int act) {
    dim3 grid(Nn / 128, (Mm + 127) / 128);
    gemm_bf16_kernel<<<grid, 256, 0, stream>>>(A, Bm, bias, oF, oB, Mm, Nn, Kk, act);
  };
  auto flash = [&](const bf16_t* Qp, int qs, const bf16_t* Kq, const bf16_t* Vq,
                   int kvs, bf16_t* Op, int os, int Sqv, int Skvv) {
    dim3 grid(Sqv / 64, B * H);
    flash_attn_kernel<<<grid, 128, 0, stream>>>(Qp, qs, Kq, Vq, kvs, Op, os,
                                                Sqv, Skvv, H, 0.125f);
  };
  auto ln = [&](const float* X, const float* Dl, const float* G, const float* Bt,
                float* Yf, bf16_t* Yb) {
    residual_ln_kernel<<<M, 256, 0, stream>>>(X, Dl, G, Bt, Yf, Yb);
  };

  // fp32 -> bf16 conversions
  cvt(x, xb, (long long)M * D);
  cvt(y, yb, (long long)MY * DC);
  cvt(sa_in_w, w_in, (long long)D * 3 * D);
  cvt(sa_ow,   w_so, (long long)D * D);
  cvt(ca_qw,   w_cq, (long long)D * D);
  cvt(ca_kw,   w_ck, (long long)DC * D);
  cvt(ca_vw,   w_cv, (long long)DC * D);
  cvt(ca_ow,   w_co, (long long)D * D);
  cvt(ff_w1,   w_f1, (long long)D * FF);
  cvt(ff_w2,   w_f2, (long long)FF * D);

  // ---- self-attention ----
  gemm(xb, w_in, sa_in_b, nullptr, qkv, M, 3 * D, D, 0);
  flash(qkv, 3 * D, qkv + D, qkv + 2 * D, 3 * D, attn, D, S, S);
  gemm(attn, w_so, sa_ob, proj, nullptr, M, D, D, 0);
  ln(x, proj, ln1_g, ln1_b, x1f, x1b);

  // ---- cross-attention ----
  gemm(x1b, w_cq, ca_qb, nullptr, qc, M, D, D, 0);
  gemm(yb, w_ck, ca_kb, nullptr, kc, MY, D, DC, 0);
  gemm(yb, w_cv, ca_vb, nullptr, vc, MY, D, DC, 0);
  flash(qc, D, kc, vc, D, attn, D, S, 77);
  gemm(attn, w_co, ca_ob, proj, nullptr, M, D, D, 0);
  ln(x1f, proj, ln2_g, ln2_b, x2f, x2b);

  // ---- feed-forward ----
  gemm(x2b, w_f1, ff_b1, nullptr, ffh, M, FF, D, 1);   // fused tanh-GELU
  gemm(ffh, w_f2, ff_b2, proj, nullptr, M, D, FF, 0);
  ln(x2f, proj, ln3_g, ln3_b, out, nullptr);
}
